// GCNEncoder_17231408791699
// MI455X (gfx1250) — matrix operand order repaired and checked by reading, not compiled, not run on hardware
//
#include <hip/hip_runtime.h>

typedef __attribute__((ext_vector_type(16))) __bf16 v16bf;
typedef __attribute__((ext_vector_type(8)))  __bf16 v8bf;
typedef __attribute__((ext_vector_type(8)))  float  v8f;

#define D 128

// ---------------- degree / normalization ----------------

__global__ void fill1_kernel(float* __restrict__ p, int n) {
  int i = blockIdx.x * blockDim.x + threadIdx.x;
  if (i < n) p[i] = 1.0f;   // self-loop contributes 1 to every node's degree
}

__global__ void deg_count_kernel(const long long* __restrict__ dst,
                                 float* __restrict__ deg, int E) {
  int i = blockIdx.x * blockDim.x + threadIdx.x;
  if (i < E) atomicAdd(&deg[(int)dst[i]], 1.0f);
}

__global__ void deg_rsqrt_kernel(float* __restrict__ p, int n) {
  int i = blockIdx.x * blockDim.x + threadIdx.x;
  if (i < n) { float d = p[i]; p[i] = (d > 0.0f) ? rsqrtf(d) : 0.0f; }
}

__global__ void zero_kernel(float* __restrict__ p, int n) {
  int i = blockIdx.x * blockDim.x + threadIdx.x;
  if (i < n) p[i] = 0.0f;
}

// ---------------- precision staging ----------------

__global__ void f32_to_bf16_kernel(const float* __restrict__ in,
                                   __bf16* __restrict__ outp, int n) {
  int i = blockIdx.x * blockDim.x + threadIdx.x;
  if (i < n) outp[i] = (__bf16)in[i];
}

// W[k][n] fp32 -> Wt[n][k] bf16 (so B fragments in LDS are contiguous per lane)
__global__ void w_to_bf16t_kernel(const float* __restrict__ W,
                                  __bf16* __restrict__ Wt) {
  int i = blockIdx.x * blockDim.x + threadIdx.x;   // 16384 threads
  int k = i >> 7, n = i & (D - 1);
  Wt[n * D + k] = (__bf16)W[k * D + n];
}

// ---------------- async global -> LDS staging (CDNA5 path, ASYNCcnt) --------

__device__ __forceinline__ void async_copy_b128(uint32_t lds_byte_addr,
                                                const void* gptr) {
  asm volatile("global_load_async_to_lds_b128 %0, %1, off"
               :: "v"(lds_byte_addr), "v"((unsigned long long)(uintptr_t)gptr)
               : "memory");
}

__device__ __forceinline__ void wait_asynccnt0() {
  asm volatile("s_wait_asynccnt 0" ::: "memory");
}

// ---------------- WMMA GEMM: H[M,128] = Xbf[M,128] @ W[128,128] -------------
// Block = 256 threads = 8 wave32. Block tile: 64 rows x 128 cols.
// Wave w owns N strip [16w,16w+16); computes 4 M sub-tiles, reusing its B frag.
// A tile (64x128 bf16, 16 KB) and Wt (128x128 bf16, 32 KB) staged via
// global_load_async_to_lds_b128; fragments read back as ds_load_b128.

__global__ void __launch_bounds__(256)
gcn_gemm_wmma(const __bf16* __restrict__ Xbf, const __bf16* __restrict__ Wt,
              float* __restrict__ H, int M) {
  __shared__ __align__(16) __bf16 sA[64 * D];    // [m][k]
  __shared__ __align__(16) __bf16 sBt[D * D];    // [n][k]
  const int tid = threadIdx.x;
  const int bm  = blockIdx.x * 64;

  const uint32_t sA_base  = (uint32_t)(uintptr_t)(void*)sA;
  const uint32_t sBt_base = (uint32_t)(uintptr_t)(void*)sBt;
  const __bf16* gA = Xbf + (size_t)bm * D;

  // 64*128 bf16 = 1024 x 16B transfers; 128*128 bf16 = 2048 x 16B transfers
  for (int i = tid; i < (64 * D) / 8; i += 256)
    async_copy_b128(sA_base + i * 16, gA + i * 8);
  for (int i = tid; i < (D * D) / 8; i += 256)
    async_copy_b128(sBt_base + i * 16, Wt + i * 8);
  wait_asynccnt0();
  __syncthreads();

  const int wave = tid >> 5;
  const int lane = tid & 31;
  const int half = lane >> 4;     // 0: lanes 0-15, 1: lanes 16-31
  const int l16  = lane & 15;
  const int n0   = wave * 16;

  v8f acc[4] = {};
  for (int kk = 0; kk < D; kk += 32) {
    // B fragment (K=32 x N=16): lane half k-range, 16 contiguous bf16
    const __bf16* bp = sBt + (n0 + l16) * D + kk + half * 16;
    v8bf b0 = *(const v8bf*)bp;
    v8bf b1 = *(const v8bf*)(bp + 8);
    v16bf bfrag = __builtin_shufflevector(b0, b1, 0, 1, 2, 3, 4, 5, 6, 7,
                                          8, 9, 10, 11, 12, 13, 14, 15);
#pragma unroll
    for (int mt = 0; mt < 4; ++mt) {
      // A fragment: element j -> K = kk + half*16 + j
      const __bf16* ap = sA + (mt * 16 + l16) * D + kk + (half << 4);
      v8bf a0 = *(const v8bf*)ap;
      v8bf a1 = *(const v8bf*)(ap + 8);
      v16bf afrag = __builtin_shufflevector(a0, a1, 0, 1, 2, 3, 4, 5, 6, 7,
                                            8, 9, 10, 11, 12, 13, 14, 15);
      acc[mt] = __builtin_amdgcn_wmma_f32_16x16x32_bf16(
          /*neg_a=*/false, afrag, /*neg_b=*/false, bfrag,
          /*c_mod=*/(short)0, acc[mt], /*reuse_a=*/false, /*reuse_b=*/false);
    }
  }

  // C/D layout: VGPR r, lanes 0-15 -> M=r, lanes 16-31 -> M=8+r; N = lane%16
#pragma unroll
  for (int mt = 0; mt < 4; ++mt) {
#pragma unroll
    for (int r = 0; r < 8; ++r) {
      int row = bm + mt * 16 + half * 8 + r;
      if (row < M) H[(size_t)row * D + n0 + l16] = acc[mt][r];
    }
  }
}

// ---------------- edge scatter: agg[dst] += dinv[src]*dinv[dst] * H[src] -----
// One wave32 per edge; each lane: float4 gather + 4 f32 atomics. H (51.2 MB)
// fits in the 192 MB L2, so gather + atomics resolve in L2, not HBM.

__global__ void __launch_bounds__(256)
scatter_edges(const float* __restrict__ H, const long long* __restrict__ src,
              const long long* __restrict__ dst, const float* __restrict__ dinv,
              float* __restrict__ agg, int E) {
  int wv   = (int)((blockIdx.x * (size_t)blockDim.x + threadIdx.x) >> 5);
  int lane = threadIdx.x & 31;
  if (wv >= E) return;
  int s = (int)src[wv];
  int d = (int)dst[wv];
  float w = dinv[s] * dinv[d];
  float4 v = ((const float4*)(H + (size_t)s * D))[lane];
  float* ap = agg + (size_t)d * D + lane * 4;
  atomicAdd(ap + 0, w * v.x);
  atomicAdd(ap + 1, w * v.y);
  atomicAdd(ap + 2, w * v.z);
  atomicAdd(ap + 3, w * v.w);
}

// ------- finalize: + self-loop + bias (+ReLU); emit bf16 acts or fp32 out ---

__global__ void finalize_kernel(const float* __restrict__ agg,
                                const float* __restrict__ H,
                                const float* __restrict__ dinv,
                                const float* __restrict__ bias,
                                float* __restrict__ outF,
                                __bf16* __restrict__ outBf,
                                int n, int relu, int emit_bf16) {
  int i = blockIdx.x * blockDim.x + threadIdx.x;
  if (i >= n) return;
  int row = i >> 7;
  int col = i & (D - 1);
  float di = dinv[row];
  float v  = agg[i] + di * di * H[i] + bias[col];
  if (relu) v = fmaxf(v, 0.0f);
  if (emit_bf16) outBf[i] = (__bf16)v;   // feeds next layer's WMMA GEMM
  else           outF[i]  = v;           // final fp32 result
}

// ---------------- host orchestration ----------------

extern "C" void kernel_launch(void* const* d_in, const int* in_sizes, int n_in,
                              void* d_out, int out_size, void* d_ws, size_t ws_size,
                              hipStream_t stream) {
  (void)n_in; (void)out_size; (void)ws_size;
  const float*     x  = (const float*)d_in[0];
  const long long* ei = (const long long*)d_in[1];   // int64 [2, E]
  const float* Wl[3] = {(const float*)d_in[2], (const float*)d_in[4], (const float*)d_in[6]};
  const float* bl[3] = {(const float*)d_in[3], (const float*)d_in[5], (const float*)d_in[7]};

  const int N  = in_sizes[0] / D;      // 100000
  const int E  = in_sizes[1] / 2;      // 1600000
  const int NE = N * D;                // 12.8M elements

  float* out = (float*)d_out;

  // ws layout (256B aligned): dinv | H (fp32) | Xbf (bf16) | Wt (bf16)
  // Wt sits right after Xbf so the last GEMM block's <=8KB A overread is safe.
  char* ws = (char*)d_ws;
  size_t o0 = 0;
  size_t o1 = (o0 + (size_t)N * 4 + 255) & ~(size_t)255;          // H
  size_t o2 = (o1 + (size_t)NE * 4 + 255) & ~(size_t)255;         // Xbf
  size_t o3 = (o2 + (size_t)NE * 2 + 255) & ~(size_t)255;         // Wt
  float*  dinv = (float*)(ws + o0);
  float*  H    = (float*)(ws + o1);
  __bf16* Xbf  = (__bf16*)(ws + o2);
  __bf16* Wt   = (__bf16*)(ws + o3);

  const long long* src = ei;
  const long long* dst = ei + E;

  // symmetric normalization: dinv = rsqrt(1 + in-degree)
  fill1_kernel<<<(N + 255) / 256, 256, 0, stream>>>(dinv, N);
  deg_count_kernel<<<(E + 255) / 256, 256, 0, stream>>>(dst, dinv, E);
  deg_rsqrt_kernel<<<(N + 255) / 256, 256, 0, stream>>>(dinv, N);

  // layer-1 activations in bf16
  f32_to_bf16_kernel<<<(NE + 255) / 256, 256, 0, stream>>>(x, Xbf, NE);

  for (int l = 0; l < 3; ++l) {
    w_to_bf16t_kernel<<<(D * D) / 256, 256, 0, stream>>>(Wl[l], Wt);
    gcn_gemm_wmma<<<(N + 63) / 64, 256, 0, stream>>>(Xbf, Wt, H, N);
    zero_kernel<<<(NE + 255) / 256, 256, 0, stream>>>(out, NE);
    scatter_edges<<<(E + 7) / 8, 256, 0, stream>>>(H, src, dst, dinv, out, E);
    finalize_kernel<<<(NE + 255) / 256, 256, 0, stream>>>(
        out, H, dinv, bl[l], out, Xbf, NE, l < 2 ? 1 : 0, l < 2 ? 1 : 0);
  }
}
